// DCINet_87711822119113
// MI455X (gfx1250) — matrix operand order repaired and checked by reading, not compiled, not run on hardware
//
#include <hip/hip_runtime.h>
#include <hip/hip_bf16.h>

#define NN 50000
#define HH 128
#define HHALF 64
#define MAXDEG 200
#define LDSROW 132   // padded fp32 row stride: 132 % 64 banks = 4 -> conflict-free fragment reads

typedef __attribute__((ext_vector_type(16))) _Float16 v16h;
typedef __attribute__((ext_vector_type(8)))  float    v8f;

// ---------------- utility kernels ----------------

__global__ void zero_i32_kernel(int* p, int n) {
  int i = blockIdx.x * blockDim.x + threadIdx.x;
  if (i < n) p[i] = 0;
}

__global__ void zero_f32_kernel(float* p, long long n) {
  long long i = (long long)blockIdx.x * blockDim.x + threadIdx.x;
  if (i < n) p[i] = 0.0f;
}

__global__ void cvt_f16_kernel(const float* __restrict__ s, _Float16* __restrict__ d, int n) {
  int i = blockIdx.x * blockDim.x + threadIdx.x;
  if (i < n) d[i] = (_Float16)s[i];
}

// degrees: out-degree on src (for embedding), in-degree on dst (for mean)
__global__ void degree_kernel(const int* __restrict__ src, const int* __restrict__ dst,
                              int* outdeg, int* indeg, int e) {
  int i = blockIdx.x * blockDim.x + threadIdx.x;
  if (i < e) {
    atomicAdd(&outdeg[src[i]], 1);
    atomicAdd(&indeg[dst[i]], 1);
  }
}

__global__ void inv_cnt_kernel(const int* __restrict__ indeg, float* __restrict__ inv, int n) {
  int i = blockIdx.x * blockDim.x + threadIdx.x;
  if (i < n) inv[i] = 1.0f / fmaxf((float)indeg[i], 1.0f);
}

// h[n][j] = relu(x[n]*W_enc[j] + b_enc[j] + deg_emb[clamp(deg,0,199)][j])
__global__ void encoder_kernel(const float* __restrict__ x, const int* __restrict__ outdeg,
                               const float* __restrict__ W_enc, const float* __restrict__ b_enc,
                               const float* __restrict__ deg_emb, float* __restrict__ h, int n) {
  long long idx = (long long)blockIdx.x * blockDim.x + threadIdx.x;
  if (idx >= (long long)n * HH) return;
  int node = (int)(idx >> 7);
  int j = (int)(idx & (HH - 1));
  int d = outdeg[node]; d = d > (MAXDEG - 1) ? (MAXDEG - 1) : d;
  float v = x[node] * W_enc[j] + b_enc[j] + deg_emb[d * HH + j];
  h[idx] = fmaxf(v, 0.0f);
}

// agg[dst[e]][j] += h[src[e]][j]  (thread = (edge, channel); coalesced rows)
__global__ void scatter_add_kernel(const int* __restrict__ src, const int* __restrict__ dst,
                                   const float* __restrict__ hin, float* __restrict__ agg, int e) {
  long long idx = (long long)blockIdx.x * blockDim.x + threadIdx.x;
  if (idx >= (long long)e * HH) return;
  int ed = (int)(idx >> 7);
  int j = (int)(idx & (HH - 1));
  atomicAdd(&agg[(long long)dst[ed] * HH + j], hin[(long long)src[ed] * HH + j]);
}

// ---------------- fused SAGE layer GEMM ----------------
// out[m][j] = relu( (agg[m]*invc[m]) @ Wl^T + bl + hin[m] @ Wr^T )[j]  (+ hin[m][j] if residual)
// Block = 256 thr (8 waves). Tile: 16 nodes x 128 cols; each wave owns one 16x16 column tile.
// A-panels (agg rows, hin rows) are staged fp32 into LDS ONCE per block via async
// global->LDS DMA (ASYNCcnt), then each wave builds f16 WMMA fragments from LDS.
__global__ __launch_bounds__(256) void sage_gemm_kernel(
    const float* __restrict__ agg, const float* __restrict__ invc,
    const float* __restrict__ hin,
    const _Float16* __restrict__ Wl, const _Float16* __restrict__ Wr,
    const float* __restrict__ bl, float* __restrict__ out,
    int nnodes, int residual) {
  __shared__ float sAgg[16 * LDSROW];
  __shared__ float sH[16 * LDSROW];

  const int tid  = threadIdx.x;
  const int wave = tid >> 5;   // 0..7 -> column tile
  const int lane = tid & 31;
  const int row0 = blockIdx.x * 16;
  const int col0 = wave * 16;
  const int r = lane & 15;

  // ---- async-stage the two 16x128 fp32 A-panels into LDS (each lane: one 16B chunk x2 iters) ----
  for (int i = tid; i < 512; i += 256) {        // 512 x 16B chunks per matrix
    int rr = i >> 5;                            // row 0..15
    int c4 = i & 31;                            // 16B chunk within row
    int node = row0 + rr;
    int nc = node < nnodes ? node : (nnodes - 1);
    const float* gA = agg + (long long)nc * HH + c4 * 4;
    const float* gH = hin + (long long)nc * HH + c4 * 4;
    unsigned la = (unsigned)(size_t)(sAgg + rr * LDSROW + c4 * 4);
    unsigned lh = (unsigned)(size_t)(sH   + rr * LDSROW + c4 * 4);
    asm volatile("global_load_async_to_lds_b128 %0, %1, off" :: "v"(la), "v"(gA) : "memory");
    asm volatile("global_load_async_to_lds_b128 %0, %1, off" :: "v"(lh), "v"(gH) : "memory");
  }
  asm volatile("s_wait_asynccnt 0x0" ::: "memory");
  __syncthreads();

  const int node = row0 + r;
  const int nc = node < nnodes ? node : (nnodes - 1);
  const float ic = invc[nc];

  const int abase = (lane < 16) ? 0 : 8;    // A-fragment K sub-offset (ISA 7.12.2)
  const int bbase = (lane < 16) ? 0 : 8;    // B-fragment K sub-offset
  const float*    sA  = sAgg + r * LDSROW;  // this lane's A row (fp32, in LDS)
  const float*    sHr = sH   + r * LDSROW;
  const _Float16* wlr = Wl + (col0 + r) * HH;  // row j of Wl == column j of B
  const _Float16* wrr = Wr + (col0 + r) * HH;

  v8f acc = {};
#pragma unroll
  for (int kc = 0; kc < HH; kc += 32) {
    v16h a, b;
    // A = mean-aggregated neighbors (scaled on the fly, LDS -> f16)
#pragma unroll
    for (int v = 0; v < 4; ++v) {
      int k0 = kc + abase + 2 * v;
      a[2 * v]     = (_Float16)(sA[k0]     * ic);
      a[2 * v + 1] = (_Float16)(sA[k0 + 1] * ic);
      int k1 = kc + 16 + abase + 2 * v;
      a[8 + 2 * v]     = (_Float16)(sA[k1]     * ic);
      a[8 + 2 * v + 1] = (_Float16)(sA[k1 + 1] * ic);
    }
#pragma unroll
    for (int v = 0; v < 8; ++v) {
      int k = kc + ((v < 4) ? 0 : 8) + bbase + 2 * v;
      b[2 * v]     = wlr[k];
      b[2 * v + 1] = wlr[k + 1];
    }
    acc = __builtin_amdgcn_wmma_f32_16x16x32_f16(false, a, false, b, (short)0, acc, false, false);

    // A = self features
#pragma unroll
    for (int v = 0; v < 4; ++v) {
      int k0 = kc + abase + 2 * v;
      a[2 * v]     = (_Float16)sHr[k0];
      a[2 * v + 1] = (_Float16)sHr[k0 + 1];
      int k1 = kc + 16 + abase + 2 * v;
      a[8 + 2 * v]     = (_Float16)sHr[k1];
      a[8 + 2 * v + 1] = (_Float16)sHr[k1 + 1];
    }
#pragma unroll
    for (int v = 0; v < 8; ++v) {
      int k = kc + ((v < 4) ? 0 : 8) + bbase + 2 * v;
      b[2 * v]     = wrr[k];
      b[2 * v + 1] = wrr[k + 1];
    }
    acc = __builtin_amdgcn_wmma_f32_16x16x32_f16(false, a, false, b, (short)0, acc, false, false);
  }

  // Epilogue: C/D layout -> lane holds column N=r, VGPR v holds row v + (lane<16?0:8)
  int ccol = col0 + r;
  float bias = bl[ccol];
  int mbase = row0 + ((lane < 16) ? 0 : 8);
#pragma unroll
  for (int v = 0; v < 8; ++v) {
    int m = mbase + v;
    if (m < nnodes) {
      float val = fmaxf(acc[v] + bias, 0.0f);
      if (residual) val += hin[(long long)m * HH + ccol];   // fp32-exact residual from global
      out[(long long)m * HH + ccol] = val;
    }
  }
}

// ---------------- head ----------------
// z[m][j] = relu(h3[m] @ Wh1^T + bh1)[j], j < 64.  4 waves/block.
__global__ __launch_bounds__(128) void head_gemm_kernel(
    const float* __restrict__ hin, const _Float16* __restrict__ W,
    const float* __restrict__ b, float* __restrict__ z, int nnodes) {
  int wave = threadIdx.x >> 5;   // 0..3
  int lane = threadIdx.x & 31;
  int row0 = blockIdx.x * 16;
  int col0 = wave * 16;
  int r = lane & 15;
  int node = row0 + r;
  int nc = node < nnodes ? node : (nnodes - 1);

  const int abase = (lane < 16) ? 0 : 8;
  const int bbase = (lane < 16) ? 0 : 8;
  const float*    hrow = hin + (long long)nc * HH;
  const _Float16* wr   = W + (col0 + r) * HH;

  v8f acc = {};
#pragma unroll
  for (int kc = 0; kc < HH; kc += 32) {
    v16h a, bfr;
#pragma unroll
    for (int v = 0; v < 4; ++v) {
      int k0 = kc + abase + 2 * v;
      a[2 * v]     = (_Float16)hrow[k0];
      a[2 * v + 1] = (_Float16)hrow[k0 + 1];
      int k1 = kc + 16 + abase + 2 * v;
      a[8 + 2 * v]     = (_Float16)hrow[k1];
      a[8 + 2 * v + 1] = (_Float16)hrow[k1 + 1];
    }
#pragma unroll
    for (int v = 0; v < 8; ++v) {
      int k = kc + ((v < 4) ? 0 : 8) + bbase + 2 * v;
      bfr[2 * v]     = wr[k];
      bfr[2 * v + 1] = wr[k + 1];
    }
    acc = __builtin_amdgcn_wmma_f32_16x16x32_f16(false, a, false, bfr, (short)0, acc, false, false);
  }

  int ccol = col0 + r;
  float bias = b[ccol];
  int mbase = row0 + ((lane < 16) ? 0 : 8);
#pragma unroll
  for (int v = 0; v < 8; ++v) {
    int m = mbase + v;
    if (m < nnodes)
      z[(long long)m * HHALF + ccol] = fmaxf(acc[v] + bias, 0.0f);
  }
}

// scores[n] = z[n] . Wh2 + bh2
__global__ void score_kernel(const float* __restrict__ z, const float* __restrict__ Wh2,
                             const float* __restrict__ bh2, float* __restrict__ out, int n) {
  int i = blockIdx.x * blockDim.x + threadIdx.x;
  if (i >= n) return;
  const float* zr = z + (long long)i * HHALF;
  float s = bh2[0];
#pragma unroll
  for (int j = 0; j < HHALF; ++j) s += zr[j] * Wh2[j];
  out[i] = s;
}

// ---------------- host launch ----------------

extern "C" void kernel_launch(void* const* d_in, const int* in_sizes, int n_in,
                              void* d_out, int out_size, void* d_ws, size_t ws_size,
                              hipStream_t stream) {
  const float* x       = (const float*)d_in[0];
  const int*   ei      = (const int*)d_in[1];
  const float* W_enc   = (const float*)d_in[2];
  const float* b_enc   = (const float*)d_in[3];
  const float* deg_emb = (const float*)d_in[4];
  const float* W1l = (const float*)d_in[5];
  const float* b1l = (const float*)d_in[6];
  const float* W1r = (const float*)d_in[7];
  const float* W2l = (const float*)d_in[8];
  const float* b2l = (const float*)d_in[9];
  const float* W2r = (const float*)d_in[10];
  const float* W3l = (const float*)d_in[11];
  const float* b3l = (const float*)d_in[12];
  const float* W3r = (const float*)d_in[13];
  const float* Wh1 = (const float*)d_in[14];
  const float* bh1 = (const float*)d_in[15];
  const float* Wh2 = (const float*)d_in[16];
  const float* bh2 = (const float*)d_in[17];

  const int n = in_sizes[0];       // 50000 nodes (x is [N,1])
  const int e = in_sizes[1] / 2;   // 800000 edges
  const int* src = ei;
  const int* dst = ei + e;

  // workspace carve-up
  char* ws = (char*)d_ws;
  size_t off = 0;
  auto carve = [&](size_t bytes) -> void* {
    void* p = ws + off;
    off += (bytes + 255) & ~(size_t)255;
    return p;
  };
  const size_t nhf = (size_t)n * HH * sizeof(float);
  float*    h0     = (float*)carve(nhf);
  float*    h1     = (float*)carve(nhf);
  float*    agg    = (float*)carve(nhf);
  float*    z      = (float*)carve((size_t)n * HHALF * sizeof(float));
  float*    invc   = (float*)carve((size_t)n * sizeof(float));
  int*      outdeg = (int*)carve((size_t)n * sizeof(int));
  int*      indeg  = (int*)carve((size_t)n * sizeof(int));
  _Float16* wf     = (_Float16*)carve((size_t)7 * HH * HH * sizeof(_Float16));
  (void)ws_size; (void)n_in; (void)out_size;

  const int B = 256;
  const int HW = HH * HH; // 16384

  // degrees
  zero_i32_kernel<<<(n + B - 1) / B, B, 0, stream>>>(outdeg, n);
  zero_i32_kernel<<<(n + B - 1) / B, B, 0, stream>>>(indeg, n);
  degree_kernel<<<(e + B - 1) / B, B, 0, stream>>>(src, dst, outdeg, indeg, e);
  inv_cnt_kernel<<<(n + B - 1) / B, B, 0, stream>>>(indeg, invc, n);

  // weight repack to f16 (tiny, L2-resident afterwards)
  cvt_f16_kernel<<<(HW + B - 1) / B, B, 0, stream>>>(W1l, wf + 0 * HW, HW);
  cvt_f16_kernel<<<(HW + B - 1) / B, B, 0, stream>>>(W1r, wf + 1 * HW, HW);
  cvt_f16_kernel<<<(HW + B - 1) / B, B, 0, stream>>>(W2l, wf + 2 * HW, HW);
  cvt_f16_kernel<<<(HW + B - 1) / B, B, 0, stream>>>(W2r, wf + 3 * HW, HW);
  cvt_f16_kernel<<<(HW + B - 1) / B, B, 0, stream>>>(W3l, wf + 4 * HW, HW);
  cvt_f16_kernel<<<(HW + B - 1) / B, B, 0, stream>>>(W3r, wf + 5 * HW, HW);
  cvt_f16_kernel<<<(HHALF * HH + B - 1) / B, B, 0, stream>>>(Wh1, wf + 6 * HW, HHALF * HH);

  // encoder
  long long nh = (long long)n * HH;
  encoder_kernel<<<(unsigned)((nh + B - 1) / B), B, 0, stream>>>(x, outdeg, W_enc, b_enc, deg_emb, h0, n);

  long long eh = (long long)e * HH;
  int tileGrid = (n + 15) / 16;

  // layer 1: h1 = relu(sage(h0))
  zero_f32_kernel<<<(unsigned)((nh + B - 1) / B), B, 0, stream>>>(agg, nh);
  scatter_add_kernel<<<(unsigned)((eh + B - 1) / B), B, 0, stream>>>(src, dst, h0, agg, e);
  sage_gemm_kernel<<<tileGrid, 256, 0, stream>>>(agg, invc, h0, wf + 0 * HW, wf + 1 * HW, b1l, h1, n, 0);

  // layer 2: h2 = relu(sage(h1)) + h1   (write into h0)
  zero_f32_kernel<<<(unsigned)((nh + B - 1) / B), B, 0, stream>>>(agg, nh);
  scatter_add_kernel<<<(unsigned)((eh + B - 1) / B), B, 0, stream>>>(src, dst, h1, agg, e);
  sage_gemm_kernel<<<tileGrid, 256, 0, stream>>>(agg, invc, h1, wf + 2 * HW, wf + 3 * HW, b2l, h0, n, 1);

  // layer 3: h3 = relu(sage(h2)) + h2   (write into h1)
  zero_f32_kernel<<<(unsigned)((nh + B - 1) / B), B, 0, stream>>>(agg, nh);
  scatter_add_kernel<<<(unsigned)((eh + B - 1) / B), B, 0, stream>>>(src, dst, h0, agg, e);
  sage_gemm_kernel<<<tileGrid, 256, 0, stream>>>(agg, invc, h0, wf + 4 * HW, wf + 5 * HW, b3l, h1, n, 1);

  // head
  head_gemm_kernel<<<tileGrid, 128, 0, stream>>>(h1, wf + 6 * HW, bh1, z, n);
  score_kernel<<<(n + B - 1) / B, B, 0, stream>>>(z, Wh2, bh2, (float*)d_out, n);
}
